// SelfAttention_48077863911658
// MI455X (gfx1250) — compile-verified, matrix operand order checked by reading, not run on hardware
//
#include <hip/hip_runtime.h>

typedef __attribute__((ext_vector_type(16))) __bf16 v16bf;
typedef __attribute__((ext_vector_type(8)))  float  v8f;
typedef __attribute__((ext_vector_type(2)))  __bf16 bf2;

constexpr int B_ = 4, S_ = 2048, D_ = 2048, H_ = 16, G_ = 4, HD_ = 128;
constexpr int MTOT = B_ * S_;              // 8192 rows
constexpr int NQ   = H_ * HD_;             // 2048
constexpr int NK   = G_ * HD_;             // 512
constexpr int NQKV = NQ + 2 * NK;          // 3072
constexpr float SCALE     = 0.08838834764831845f;  // 1/sqrt(128)
constexpr float LOG_THETA = 9.210340371976184f;    // ln(10000)

// 16-bit A/B fragment K-index for v_wmma_*_16x16x32 (ISA 7.12.2):
// VGPR p holds K pair {2p, 2p+1} for p<4, {8+2p, 9+2p} for p>=4; lanes 16-31 add +8.
__device__ __forceinline__ int kidx(int e, int half) {
  int p = e >> 1;
  return (p < 4 ? 2 * p : 2 * p + 8) + (e & 1) + 8 * half;
}

// ------------------------------------------------- async global->LDS helpers
typedef __attribute__((address_space(3))) const char lds_char;
__device__ __forceinline__ unsigned lds_off(const void* p) {
  return (unsigned)(unsigned long long)(lds_char*)p;  // ptrtoint of LDS ptr = offset
}
// GLOBAL_LOAD_ASYNC_TO_LDS_B128 (GV mode): per-lane LDS[vdst] <= MEM[vaddr], ASYNCcnt++
__device__ __forceinline__ void async_copy_b128(const void* gaddr, void* laddr) {
  asm volatile("global_load_async_to_lds_b128 %0, %1, off"
               :: "v"(lds_off(laddr)), "v"(gaddr)
               : "memory");
}
__device__ __forceinline__ void wait_async0() { asm volatile("s_wait_asynccnt 0x0" ::: "memory"); }
__device__ __forceinline__ void wait_async6() { asm volatile("s_wait_asynccnt 0x6" ::: "memory"); }
__device__ __forceinline__ void wait_async8() { asm volatile("s_wait_asynccnt 0x8" ::: "memory"); }

// ------------------------------------------------- VALU cross-lane reductions
// xor-mask 1/2/4: intra-8-lane permutation -> v_mov_b32_dpp8
// xor-mask 8: intra-16-lane permutation   -> v_permlane16_b32 (constant selects)
__device__ __forceinline__ int   f2i(float x) { return __builtin_bit_cast(int, x); }
__device__ __forceinline__ float i2f(int x)   { return __builtin_bit_cast(float, x); }

constexpr int dpp8_xor_sel(int m) {
  int s = 0;
  for (int i = 0; i < 8; i++) s |= ((i ^ m) & 7) << (3 * i);
  return s;
}

template <int M>
__device__ __forceinline__ float xor_swap(float x) {
  if constexpr (M < 8) {
    return i2f(__builtin_amdgcn_mov_dpp8(f2i(x), dpp8_xor_sel(M)));
  } else {
    // lane -> lane^8 within each 16-lane row: lanes0-7 sel 8..15, lanes8-15 sel 0..7
    return i2f(__builtin_amdgcn_permlane16(f2i(x), f2i(x),
                                           (int)0xFEDCBA98u, (int)0x76543210u,
                                           true, false));
  }
}
__device__ __forceinline__ float rowmax16(float x) {
  x = fmaxf(x, xor_swap<1>(x));
  x = fmaxf(x, xor_swap<2>(x));
  x = fmaxf(x, xor_swap<4>(x));
  x = fmaxf(x, xor_swap<8>(x));
  return x;
}
__device__ __forceinline__ float rowsum16(float x) {
  x += xor_swap<1>(x);
  x += xor_swap<2>(x);
  x += xor_swap<4>(x);
  x += xor_swap<8>(x);
  return x;
}

// ---------------------------------------------------------------- conversions
__global__ void cvt_f32_bf16(__bf16* __restrict__ dst, const float* __restrict__ src, long n) {
  long i = (long)blockIdx.x * blockDim.x + threadIdx.x;
  if (i < n) dst[i] = (__bf16)src[i];
}

__global__ void pack_wqkv(__bf16* __restrict__ dst, const float* __restrict__ Wq,
                          const float* __restrict__ Wk, const float* __restrict__ Wv) {
  long i = (long)blockIdx.x * blockDim.x + threadIdx.x;
  if (i >= (long)D_ * NQKV) return;
  int j = (int)(i % NQKV);
  long d = i / NQKV;
  float v;
  if (j < NQ)            v = Wq[d * NQ + j];
  else if (j < NQ + NK)  v = Wk[d * NK + (j - NQ)];
  else                   v = Wv[d * NK + (j - NQ - NK)];
  dst[i] = (__bf16)v;
}

__global__ void pack_bias(float* __restrict__ dst, const float* __restrict__ bq,
                          const float* __restrict__ bk, const float* __restrict__ bv) {
  int j = blockIdx.x * blockDim.x + threadIdx.x;
  if (j >= NQKV) return;
  float v;
  if (j < NQ)            v = bq[j];
  else if (j < NQ + NK)  v = bk[j - NQ];
  else                   v = bv[j - NQ - NK];
  dst[j] = v;
}

// ---------------------------------------------------------------- WMMA GEMM
// C[M,N] = A[M,K](bf16) * B[K,N](bf16) + bias[N]
// Block: 128 thr = 4 waves. Tile: BM=64, BN=128, KC=32. Wave w -> 16 rows x 128 cols.
// Double-buffered async global->LDS staging: 6 b128 async ops / thread / k-tile
// (A 64x32 -> 2 chunks, B 32x128 -> 4 chunks of 8 bf16 each).
template <int MODE>
__global__ __launch_bounds__(128) void gemm_wmma(
    const __bf16* __restrict__ A, const __bf16* __restrict__ Bw,
    const float* __restrict__ bias, int M, int N, int K,
    __bf16* __restrict__ outQ, __bf16* __restrict__ outK, __bf16* __restrict__ outV,
    float* __restrict__ outF) {
  __shared__ __bf16 As[2][64 * 32];
  __shared__ __bf16 Bs[2][32 * 128];
  const int t = threadIdx.x;
  const int lane = t & 31, wave = t >> 5;
  const int half = lane >> 4, colL = lane & 15;
  const int mbase = blockIdx.y * 64;
  const int nbase = blockIdx.x * 128;

  v8f acc[8] = {};

  auto issue_tile = [&](int ks, int buf) {
#pragma unroll
    for (int i = 0; i < 2; i++) {           // A: 256 chunks of 8 elems
      int ca = i * 128 + t;
      int row = ca >> 2, c8 = (ca & 3) * 8;
      async_copy_b128(&A[(size_t)(mbase + row) * K + ks + c8], &As[buf][row * 32 + c8]);
    }
#pragma unroll
    for (int i = 0; i < 4; i++) {           // B: 512 chunks of 8 elems
      int cb = i * 128 + t;
      int row = cb >> 4, c8 = (cb & 15) * 8;
      async_copy_b128(&Bw[(size_t)(ks + row) * N + nbase + c8], &Bs[buf][row * 128 + c8]);
    }
  };

  const int niter = K / 32;
  issue_tile(0, 0);
  for (int it = 0; it < niter; ++it) {
    const int buf = it & 1;
    __syncthreads();                 // all waves done reading buf^1 (prev iter)
    if (it + 1 < niter) {
      issue_tile((it + 1) * 32, buf ^ 1);
      wait_async6();                 // in-order: tile `it` complete, next may fly
    } else {
      wait_async0();
    }
    __syncthreads();                 // tile `it` visible to all waves

    v16bf af;
#pragma unroll
    for (int p = 0; p < 8; p++) {
      bf2 pr = *(const bf2*)&As[buf][(wave * 16 + colL) * 32 + kidx(2 * p, half)];
      af[2 * p] = pr.x; af[2 * p + 1] = pr.y;
    }
#pragma unroll
    for (int nf = 0; nf < 8; nf++) {
      v16bf bfg;
#pragma unroll
      for (int e = 0; e < 16; e++)
        bfg[e] = Bs[buf][kidx(e, half) * 128 + nf * 16 + colL];
      acc[nf] = __builtin_amdgcn_wmma_f32_16x16x32_bf16(
          false, af, false, bfg, (short)0, acc[nf], false, false);
    }
  }

#pragma unroll
  for (int nf = 0; nf < 8; nf++) {
#pragma unroll
    for (int r = 0; r < 8; r++) {
      int row = mbase + wave * 16 + r + 8 * half;
      int col = nbase + nf * 16 + colL;
      float v = acc[nf][r] + bias[col];
      if (MODE == 0) {
        if (col < NQ)            outQ[(size_t)row * NQ + col] = (__bf16)v;
        else if (col < NQ + NK)  outK[(size_t)row * NK + (col - NQ)] = (__bf16)v;
        else                     outV[(size_t)row * NK + (col - NQ - NK)] = (__bf16)v;
      } else {
        outF[(size_t)row * N + col] = v;
      }
    }
  }
}

// ---------------------------------------------------------------- RoPE (in place)
__global__ void rope_kernel(__bf16* __restrict__ Qb, __bf16* __restrict__ Kb) {
  const long n = (long)MTOT * (H_ + G_) * 64;
  long idx = (long)blockIdx.x * blockDim.x + threadIdx.x;
  if (idx >= n) return;
  int i = (int)(idx & 63);
  long u = idx >> 6;
  int hg = (int)(u % (H_ + G_));
  long rs = u / (H_ + G_);
  int s = (int)(rs % S_);
  float inv = __expf(-((float)(2 * i) * (1.0f / 128.0f)) * LOG_THETA);
  float ang = (float)s * inv;
  float c = __cosf(ang), sn = __sinf(ang);
  if (hg < H_) {
    size_t base = (size_t)rs * NQ + hg * 128 + i;
    float a = (float)Qb[base], b = (float)Qb[base + 64];
    Qb[base]      = (__bf16)((a * c - b * sn) * SCALE);   // fold 1/sqrt(HD) into Q
    Qb[base + 64] = (__bf16)((b * c + a * sn) * SCALE);
  } else {
    size_t base = (size_t)rs * NK + (hg - H_) * 128 + i;
    float a = (float)Kb[base], b = (float)Kb[base + 64];
    Kb[base]      = (__bf16)(a * c - b * sn);
    Kb[base + 64] = (__bf16)(b * c + a * sn);
  }
}

// ---------------------------------------------------------------- flash attention
// Block = 128 thr = 4 waves; block handles 64 query rows of one (b,h); wave = 16 rows.
// KV streamed in 32-key tiles through double-buffered LDS via async loads.
__global__ __launch_bounds__(128) void flash_attn(
    const __bf16* __restrict__ Qb, const __bf16* __restrict__ Kb,
    const __bf16* __restrict__ Vb, __bf16* __restrict__ attn) {
  __shared__ __bf16 Ks[2][32 * 128];
  __shared__ __bf16 Vs[2][32 * 128];
  __shared__ __bf16 Ps[4][16 * 32];
  const int t = threadIdx.x;
  const int lane = t & 31, wave = t >> 5;
  const int half = lane >> 4, colL = lane & 15;
  const int b = blockIdx.y / H_, h = blockIdx.y % H_, g = h / (H_ / G_);
  const int q0 = blockIdx.x * 64;

  // Q rows for this wave as 4 A-fragments (K-dim 128 = 4 chunks of 32)
  const size_t qrowbase = (size_t)(b * S_ + q0 + wave * 16 + colL) * NQ + h * 128;
  v16bf aq[4];
#pragma unroll
  for (int kc = 0; kc < 4; kc++)
#pragma unroll
    for (int p = 0; p < 8; p++) {
      bf2 pr = *(const bf2*)&Qb[qrowbase + kc * 32 + kidx(2 * p, half)];
      aq[kc][2 * p] = pr.x; aq[kc][2 * p + 1] = pr.y;
    }

  float m[8], l[8];
  v8f o[8] = {};
#pragma unroll
  for (int r = 0; r < 8; r++) { m[r] = -1e30f; l[r] = 0.0f; }

  const size_t kvbase = (size_t)(b * S_) * NK + (size_t)g * 128;
  auto issue_kv = [&](int jb, int buf) {             // 8 async ops / thread / tile
#pragma unroll
    for (int i = 0; i < 4; i++) {
      int cid = i * 128 + t;          // 512 b128 chunks per 32x128 tile
      int row = cid >> 4;             // key within tile
      int c8  = (cid & 15) * 8;       // dim offset
      size_t gidx = kvbase + (size_t)(jb + row) * NK + c8;
      async_copy_b128(&Kb[gidx], &Ks[buf][row * 128 + c8]);
      async_copy_b128(&Vb[gidx], &Vs[buf][row * 128 + c8]);
    }
  };

  const int jend = q0 + 64;           // causal: keys up to q0+63
  const int ntile = jend / 32;
  issue_kv(0, 0);
  for (int it = 0; it < ntile; ++it) {
    const int jb = it * 32;
    const int buf = it & 1;
    __syncthreads();                  // all waves done reading buf^1 (prev iter)
    if (it + 1 < ntile) {
      issue_kv(jb + 32, buf ^ 1);
      wait_async8();                  // tile `it` complete; next tile in flight
    } else {
      wait_async0();
    }
    __syncthreads();                  // tile `it` visible block-wide

    // scores: 16 rows x 32 keys (2 col-groups), accumulate over dim=128
    v8f sc[2] = {};
#pragma unroll
    for (int cg = 0; cg < 2; cg++) {
#pragma unroll
      for (int kc = 0; kc < 4; kc++) {
        v16bf bk;  // B = K^T : element[k=dim][n=key] = Ks[key][dim]
#pragma unroll
        for (int p = 0; p < 8; p++) {
          bf2 pr = *(const bf2*)&Ks[buf][(cg * 16 + colL) * 128 + kc * 32 + kidx(2 * p, half)];
          bk[2 * p] = pr.x; bk[2 * p + 1] = pr.y;
        }
        sc[cg] = __builtin_amdgcn_wmma_f32_16x16x32_bf16(
            false, aq[kc], false, bk, (short)0, sc[cg], false, false);
      }
    }

    // causal mask + online softmax (rows live per half-wave; VALU xor-reductions)
#pragma unroll
    for (int r = 0; r < 8; r++) {
      int rowg = q0 + wave * 16 + r + 8 * half;
#pragma unroll
      for (int cg = 0; cg < 2; cg++) {
        int keyg = jb + cg * 16 + colL;
        if (keyg > rowg) sc[cg][r] = -1e30f;
      }
      float mx = rowmax16(fmaxf(sc[0][r], sc[1][r]));
      float nm = fmaxf(m[r], mx);
      float alpha = __expf(m[r] - nm);
      float sum = 0.0f;
#pragma unroll
      for (int cg = 0; cg < 2; cg++) {
        sc[cg][r] = __expf(sc[cg][r] - nm);
        sum += sc[cg][r];
      }
      sum = rowsum16(sum);
      l[r] = l[r] * alpha + sum;
      m[r] = nm;
#pragma unroll
      for (int f = 0; f < 8; f++) o[f][r] *= alpha;
    }

    // transpose P (C-layout -> A-layout) through per-wave LDS staging
#pragma unroll
    for (int r = 0; r < 8; r++)
#pragma unroll
      for (int cg = 0; cg < 2; cg++)
        Ps[wave][(r + 8 * half) * 32 + cg * 16 + colL] = (__bf16)sc[cg][r];
    __syncthreads();
    v16bf pa;
#pragma unroll
    for (int p = 0; p < 8; p++) {
      bf2 pr = *(const bf2*)&Ps[wave][colL * 32 + kidx(2 * p, half)];
      pa[2 * p] = pr.x; pa[2 * p + 1] = pr.y;
    }
    // O += P @ V  (one 16x16x32 WMMA per 16-dim column group)
#pragma unroll
    for (int f = 0; f < 8; f++) {
      v16bf bvf;  // B element[k=key][n=dim] = Vs[key][dim]
#pragma unroll
      for (int e = 0; e < 16; e++)
        bvf[e] = Vs[buf][kidx(e, half) * 128 + f * 16 + colL];
      o[f] = __builtin_amdgcn_wmma_f32_16x16x32_bf16(
          false, pa, false, bvf, (short)0, o[f], false, false);
    }
  }

#pragma unroll
  for (int f = 0; f < 8; f++)
#pragma unroll
    for (int r = 0; r < 8; r++) {
      int rowg = b * S_ + q0 + wave * 16 + r + 8 * half;
      float v = o[f][r] / l[r];
      attn[(size_t)rowg * NQ + h * 128 + f * 16 + colL] = (__bf16)v;
    }
}

// ---------------------------------------------------------------- launch
extern "C" void kernel_launch(void* const* d_in, const int* in_sizes, int n_in,
                              void* d_out, int out_size, void* d_ws, size_t ws_size,
                              hipStream_t stream) {
  (void)in_sizes; (void)n_in; (void)out_size; (void)ws_size;
  const float* x  = (const float*)d_in[0];
  const float* Wq = (const float*)d_in[1];
  const float* bq = (const float*)d_in[2];
  const float* Wk = (const float*)d_in[3];
  const float* bk = (const float*)d_in[4];
  const float* Wv = (const float*)d_in[5];
  const float* bv = (const float*)d_in[6];
  const float* Wo = (const float*)d_in[7];
  const float* bo = (const float*)d_in[8];

  char* w = (char*)d_ws;
  size_t off = 0;
  auto alloc = [&](size_t bytes) -> void* {
    void* p = w + off;
    off = (off + bytes + 255) & ~(size_t)255;
    return p;
  };
  __bf16* xb   = (__bf16*)alloc((size_t)MTOT * D_ * 2);
  __bf16* wqkv = (__bf16*)alloc((size_t)D_ * NQKV * 2);
  __bf16* wob  = (__bf16*)alloc((size_t)NQ * D_ * 2);
  float*  bqkv = (float*) alloc((size_t)NQKV * 4);
  __bf16* Qb   = (__bf16*)alloc((size_t)MTOT * NQ * 2);
  __bf16* Kb   = (__bf16*)alloc((size_t)MTOT * NK * 2);
  __bf16* Vb   = (__bf16*)alloc((size_t)MTOT * NK * 2);
  __bf16* attn = (__bf16*)alloc((size_t)MTOT * NQ * 2);

  long nx = (long)MTOT * D_;
  cvt_f32_bf16<<<dim3((unsigned)((nx + 255) / 256)), 256, 0, stream>>>(xb, x, nx);
  long nwo = (long)NQ * D_;
  cvt_f32_bf16<<<dim3((unsigned)((nwo + 255) / 256)), 256, 0, stream>>>(wob, Wo, nwo);
  long nwq = (long)D_ * NQKV;
  pack_wqkv<<<dim3((unsigned)((nwq + 255) / 256)), 256, 0, stream>>>(wqkv, Wq, Wk, Wv);
  pack_bias<<<dim3((NQKV + 255) / 256), 256, 0, stream>>>(bqkv, bq, bk, bv);

  gemm_wmma<0><<<dim3(NQKV / 128, MTOT / 64), 128, 0, stream>>>(
      xb, wqkv, bqkv, MTOT, NQKV, D_, Qb, Kb, Vb, nullptr);

  long nr = (long)MTOT * (H_ + G_) * 64;
  rope_kernel<<<dim3((unsigned)((nr + 255) / 256)), 256, 0, stream>>>(Qb, Kb);

  flash_attn<<<dim3(S_ / 64, B_ * H_), 128, 0, stream>>>(Qb, Kb, Vb, attn);

  gemm_wmma<1><<<dim3(D_ / 128, MTOT / 64), 128, 0, stream>>>(
      attn, wob, bo, MTOT, D_, NQ, nullptr, nullptr, nullptr, (float*)d_out);
}
